// GATv2Attention_64046552318057
// MI455X (gfx1250) — compile-verified, matrix operand order checked by reading, not run on hardware
//
#include <hip/hip_runtime.h>
#include <hip/hip_bf16.h>

typedef __attribute__((ext_vector_type(16))) _Float16 v16h;
typedef __attribute__((ext_vector_type(8)))  float    v8f;

// Problem constants (from reference setup_inputs)
constexpr int B  = 2;
constexpr int H  = 8;
constexpr int LQ = 512;
constexpr int LK = 512;
constexpr int D  = 32;
constexpr int QT = 16;          // q rows per tile (block)
constexpr int KC = 16;          // k rows per chunk (one WMMA column group)
constexpr int NW = 4;           // waves per block (Lk split 4 ways)
constexpr int CHUNKS_PER_WAVE = (LK / KC) / NW;   // 8

__device__ __forceinline__ float fast_exp2(float x) {
#if __has_builtin(__builtin_amdgcn_exp2f)
    return __builtin_amdgcn_exp2f(x);
#else
    return exp2f(x);
#endif
}

__device__ __forceinline__ float fast_rcp(float x) {
#if __has_builtin(__builtin_amdgcn_rcpf)
    return __builtin_amdgcn_rcpf(x);
#else
    return 1.0f / x;
#endif
}

__device__ __forceinline__ float silu_f32(float x) {
#if __has_builtin(__builtin_amdgcn_tanhf)
    // sigmoid(x) = 0.5*tanh(x/2) + 0.5  -> single gfx1250 transcendental
    return x * (0.5f * __builtin_amdgcn_tanhf(0.5f * x) + 0.5f);
#else
    return x * fast_rcp(1.0f + fast_exp2(-1.44269504f * x));
#endif
}

// Compiler ordering fence + LDS completion wait (same-wave DS ops are
// architecturally in-order; the s_wait_dscnt is belt-and-braces).
__device__ __forceinline__ void wave_lds_fence() {
    __builtin_amdgcn_wave_barrier();
    asm volatile("s_wait_dscnt 0x0" ::: "memory");
}

__global__ __launch_bounds__(NW * 32)
void GATv2Attention_64046552318057_kernel(const float* __restrict__ q,
                                          const float* __restrict__ k,
                                          const unsigned char* __restrict__ mask,
                                          const float* __restrict__ bias,
                                          const float* __restrict__ att,
                                          float* __restrict__ out) {
    __shared__ __align__(16) float sk[NW][KC * D];  // per-wave k staging (4x2 KB)
    __shared__ __align__(16) float sc[QT * LK];     // score rows        (32 KB)

    const int tid  = threadIdx.x;
    const int wave = tid >> 5;
    const int lane = tid & 31;

    const int tile = blockIdx.x;                    // B*H*(LQ/QT) tiles
    const int qt   = tile & (LQ / QT - 1);
    const int bh   = tile >> 5;                     // (LQ/QT)==32
    const int b    = bh >> 3;                       // /H
    const int h    = bh & (H - 1);
    const int q0   = qt * QT;

    const int aRow  = lane & 15;                    // A row / C column / my B column
    const int hiK   = (lane >> 4) << 3;             // K half select (0 or 8)
    const int cMoff = (lane >> 4) << 3;             // C M-offset (0 or 8)

    // ---- per-tile loads: q row fragment + attention fragment (f16) ----
    float qv[16];
    const float* qrow = q + (((size_t)(b * H + h) * LQ) + (q0 + aRow)) * D;
    const float* ah   = att + h * D;
    v16h afrag;
#pragma unroll
    for (int e = 0; e < 16; ++e) {
        const int kf = ((e < 8) ? e : e + 8) + hiK; // K index for element e
        qv[e]    = qrow[kf];
        afrag[e] = (_Float16)ah[kf];
    }
    v16h zfrag = {};

    const float* kbh = k    + ((size_t)(b * H + h) * LK) * D;
    const float* bbh = bias + ((size_t)(b * H + h) * LQ) * LK;
    float* skw = sk[wave];

    // ---- score tiles: each wave owns 8 chunks (128 k-columns) ----
    const int kcBeg = wave * CHUNKS_PER_WAVE;
    for (int kc = kcBeg; kc < kcBeg + CHUNKS_PER_WAVE; ++kc) {
        const int k0 = kc * KC;

        // stage k chunk (16 rows x 32 d = 512 floats) into LDS, 4x float4/lane
        const float* kchunk = kbh + (size_t)k0 * D;
#pragma unroll
        for (int t = 0; t < 4; ++t) {
            const int f = lane * 16 + t * 4;
            *(float4*)&skw[f] = *(const float4*)&kchunk[f];
        }
        // prefetch this wave's next chunk (global_prefetch_b8)
        if (kc + 1 < kcBeg + CHUNKS_PER_WAVE)
            __builtin_prefetch((const char*)(kbh + (size_t)(k0 + KC) * D) + lane * 64, 0, 0);
        wave_lds_fence();   // order staging stores before the silu loads

        // C init from attn_bias tile (C layout: M = r + cMoff, N = aRow)
        v8f c;
        const float* bp = bbh + (size_t)(q0 + cMoff) * LK + k0 + aRow;
#pragma unroll
        for (int r = 0; r < 8; ++r) c[r] = bp[(size_t)r * LK];

        // 16 accumulating WMMAs: column j of the tile comes from B_j = a (x) e_j
#pragma unroll 2
        for (int j = 0; j < KC; ++j) {
            v16h afr;
#pragma unroll
            for (int e = 0; e < 16; ++e) {
                const int kf = ((e < 8) ? e : e + 8) + hiK;
                afr[e] = (_Float16)silu_f32(qv[e] + skw[j * D + kf]);
            }
            const v16h bfr = (j == aRow) ? afrag : zfrag;
            c = __builtin_amdgcn_wmma_f32_16x16x32_f16(
                    false, afr, false, bfr, (short)0, c, false, false);
        }

        // scatter tile into score buffer (this wave's private columns)
#pragma unroll
        for (int r = 0; r < 8; ++r)
            sc[(cMoff + r) * LK + k0 + aRow] = c[r];
        wave_lds_fence();   // order tile stores before next chunk's staging
    }
    __syncthreads();        // all 4 waves' score columns visible

    // ---- masked softmax over LK: wave w handles rows [4w, 4w+4) ----
    const unsigned char* mb = mask + ((size_t)b * LQ + q0) * LK;
    for (int r = wave * (QT / NW); r < (wave + 1) * (QT / NW); ++r) {
        float* srow = &sc[r * LK];
        const unsigned char* mr = mb + (size_t)r * LK;

        float mx = -INFINITY;
        for (int t = lane; t < LK; t += 32) {
            float v = srow[t];
            if (mr[t]) v = -INFINITY;
            srow[t] = v;
            mx = fmaxf(mx, v);
        }
#pragma unroll
        for (int off = 16; off; off >>= 1)
            mx = fmaxf(mx, __shfl_xor(mx, off, 32));

        float* orow = out + (((size_t)(b * H + h) * LQ) + (q0 + r)) * LK;
        if (mx == -INFINITY) {
            // fully masked row -> nan_to_num(0)
            for (int t = lane; t < LK; t += 32) orow[t] = 0.0f;
            continue;
        }

        float pbuf[LK / 32];
        float sum = 0.0f;
        for (int t = lane, i = 0; t < LK; t += 32, ++i) {
            const float p = fast_exp2((srow[t] - mx) * 1.44269504f);
            pbuf[i] = p;
            sum += p;
        }
#pragma unroll
        for (int off = 16; off; off >>= 1)
            sum += __shfl_xor(sum, off, 32);

        const float inv = fast_rcp(sum);
        for (int t = lane, i = 0; t < LK; t += 32, ++i)
            orow[t] = pbuf[i] * inv;
    }
}

extern "C" void kernel_launch(void* const* d_in, const int* in_sizes, int n_in,
                              void* d_out, int out_size, void* d_ws, size_t ws_size,
                              hipStream_t stream) {
    const float*         q    = (const float*)d_in[0];
    const float*         k    = (const float*)d_in[1];
    /* d_in[2] = scale (unused by the module) */
    const unsigned char* mask = (const unsigned char*)d_in[3];
    const float*         bias = (const float*)d_in[4];
    const float*         att  = (const float*)d_in[5];
    float*               out  = (float*)d_out;

    const int nTiles = B * H * (LQ / QT);   // 512 blocks x 4 waves = 2048 waves
    GATv2Attention_64046552318057_kernel<<<nTiles, NW * 32, 0, stream>>>(
        q, k, mask, bias, att, out);
}
